// MultiHeadedAttention_53678501265881
// MI455X (gfx1250) — matrix operand order repaired and checked by reading, not compiled, not run on hardware
//
#include <hip/hip_runtime.h>
#include <hip/hip_bf16.h>

// ---------------------------------------------------------------------------
// MHA forward for MI455X (gfx1250, wave32, WMMA 16x16x32 f16 / f32-acc).
//   B=2, S=2048, D=1024, H=16, DK=64
// Stages:
//   1) gemm_qkvo<float,float,true> : fused Q/K/V projections (gridDim.z=3),
//                                    f32 -> f16 converted in LDS staging,
//                                    head-split f16 output [B,H,S,DK] in d_ws.
//   2) flash_attn                  : streaming softmax(QK^T/8 + causal)·V with
//                                    all matmuls on v_wmma; writes f16 context
//                                    [B,S,D] + attn_weights (mean over d_k).
//   3) gemm_qkvo<_Float16,float,false>: output projection, f32 into d_out.
// Workspace: 4 x 8MB f16 arrays (q,k,v,ctx) = 32MB of d_ws.
// ---------------------------------------------------------------------------

typedef __attribute__((ext_vector_type(16))) _Float16 v16h;
typedef __attribute__((ext_vector_type(8)))  _Float16 v8h;
typedef __attribute__((ext_vector_type(8)))  float    v8f;

constexpr int Bb  = 2;
constexpr int Ss  = 2048;
constexpr int Dd  = 1024;
constexpr int Hh  = 16;
constexpr int DKk = 64;
constexpr int Mrows = Bb * Ss;   // 4096 token rows

__device__ inline v16h mk16(v8h lo, v8h hi) {
  v16h r;
#pragma unroll
  for (int i = 0; i < 8; ++i) { r[i] = lo[i]; r[i + 8] = hi[i]; }
  return r;
}

// Load 16 contiguous input elements as 16 halves (convert if f32 source).
template <typename T>
__device__ inline void load16h(const T* __restrict__ g, v8h& h0, v8h& h1) {
  if constexpr (sizeof(T) == 4) {
    const float4* gp = (const float4*)g;
    float4 f0 = gp[0], f1 = gp[1], f2 = gp[2], f3 = gp[3];
    h0 = (v8h){(_Float16)f0.x, (_Float16)f0.y, (_Float16)f0.z, (_Float16)f0.w,
               (_Float16)f1.x, (_Float16)f1.y, (_Float16)f1.z, (_Float16)f1.w};
    h1 = (v8h){(_Float16)f2.x, (_Float16)f2.y, (_Float16)f2.z, (_Float16)f2.w,
               (_Float16)f3.x, (_Float16)f3.y, (_Float16)f3.z, (_Float16)f3.w};
  } else {
    const v8h* gp = (const v8h*)g;
    h0 = gp[0];
    h1 = gp[1];
  }
}

// ---------------------------------------------------------------------------
// Tiled WMMA GEMM:  out[m,n] = sum_k A[m,k] * W[n,k] + bias[n]
// Block = 128 threads (4 wave32), 64x64 output tile, K-step 32.
// blockIdx.z selects one of three (A, W, bias, out) tuples (fused QKV).
// TA = activation dtype, TW = weight dtype (both converted to f16 in LDS).
// HEADSPLIT: write f16 into [B,H,S,DK]; else write f32 row-major [M,N].
// ---------------------------------------------------------------------------
template <typename TA, typename TW, bool HEADSPLIT>
__global__ __launch_bounds__(128) void gemm_qkvo(
    const TA* __restrict__ A0, const TA* __restrict__ A1, const TA* __restrict__ A2,
    const TW* __restrict__ W0, const TW* __restrict__ W1, const TW* __restrict__ W2,
    const float* __restrict__ b0, const float* __restrict__ b1, const float* __restrict__ b2,
    void* __restrict__ o0, void* __restrict__ o1, void* __restrict__ o2) {
  const int z = blockIdx.z;
  const TA*    A    = (z == 0) ? A0 : (z == 1) ? A1 : A2;
  const TW*    W    = (z == 0) ? W0 : (z == 1) ? W1 : W2;
  const float* bias = (z == 0) ? b0 : (z == 1) ? b1 : b2;
  void*        out  = (z == 0) ? o0 : (z == 1) ? o1 : o2;

  const int n0 = blockIdx.x * 64;
  const int m0 = blockIdx.y * 64;

  __shared__ __align__(16) _Float16 As[64][40];   // [row][k] pad->40 halves
  __shared__ __align__(16) _Float16 Bs[64][40];   // [n  ][k] (= W row-major)

  const int t    = threadIdx.x;
  const int wave = t >> 5;
  const int lane = t & 31;
  const int lm   = lane & 15;   // fragment row/col within 16
  const int hiq  = lane >> 4;   // half-wave select

  v8f acc[4] = {};              // 4 x (16x16) f32 accumulators per wave

  const int r    = t >> 1;            // staged tile row (0..63)
  const int koff = (t & 1) * 16;      // 16-element chunk within K-step

  for (int k0 = 0; k0 < Dd; k0 += 32) {
    __syncthreads();
    {
      const TA* ga = A + (size_t)(m0 + r) * Dd + k0 + koff;
      v8h h0, h1;
      load16h(ga, h0, h1);
      *(v8h*)&As[r][koff]     = h0;
      *(v8h*)&As[r][koff + 8] = h1;

      const TW* gw = W + (size_t)(n0 + r) * Dd + k0 + koff;
      load16h(gw, h0, h1);
      *(v8h*)&Bs[r][koff]     = h0;
      *(v8h*)&Bs[r][koff + 8] = h1;

      if (k0 + 32 < Dd) {                       // hint next K tile into GL2
        __builtin_prefetch(ga + 32, 0, 0);      // -> global_prefetch_b8
        __builtin_prefetch(gw + 32, 0, 0);
      }
    }
    __syncthreads();

    // A fragment: 16x32, this wave's 16 rows.
    v16h af = mk16(*(const v8h*)&As[wave * 16 + lm][hiq * 8],
                   *(const v8h*)&As[wave * 16 + lm][16 + hiq * 8]);
#pragma unroll
    for (int tn = 0; tn < 4; ++tn) {
      v16h bf = mk16(*(const v8h*)&Bs[tn * 16 + lm][hiq * 8],
                     *(const v8h*)&Bs[tn * 16 + lm][16 + hiq * 8]);
      acc[tn] = __builtin_amdgcn_wmma_f32_16x16x32_f16(
          false, af, false, bf, (short)0, acc[tn], false, false);
    }
  }

  // Epilogue: add bias, write out.
#pragma unroll
  for (int tn = 0; tn < 4; ++tn) {
    const int n  = n0 + tn * 16 + lm;
    const float bval = bias[n];
#pragma unroll
    for (int j = 0; j < 8; ++j) {
      const int m = m0 + wave * 16 + j + 8 * hiq;   // C layout: VGPR j -> M=j(+8)
      const float val = acc[tn][j] + bval;
      if constexpr (HEADSPLIT) {
        const int bb = m / Ss, s = m % Ss;
        const int hh = n >> 6, dk = n & 63;
        ((_Float16*)out)[(((size_t)(bb * Hh + hh)) * Ss + s) * DKk + dk] =
            (_Float16)val;
      } else {
        ((float*)out)[(size_t)m * Dd + n] = val;
      }
    }
  }
}

// ---------------------------------------------------------------------------
// Flash attention: block = (b*H+h, 64-query-row tile), 4 wave32 waves,
// each wave owns 16 query rows. 32-key blocks of K and V^T staged in LDS.
// Causal mask applied structurally (== reference tril mask, -1e9 fill).
// ---------------------------------------------------------------------------
__global__ __launch_bounds__(128) void flash_attn(
    const _Float16* __restrict__ qp, const _Float16* __restrict__ kp,
    const _Float16* __restrict__ vp, _Float16* __restrict__ ctx,
    float* __restrict__ attw) {
  const int bh  = blockIdx.x;           // b*H + h
  const int b   = bh >> 4;
  const int h   = bh & 15;
  const int blk = blockIdx.y;           // 64-row query tile index
  const int t    = threadIdx.x;
  const int wave = t >> 5;
  const int lane = t & 31;
  const int lm   = lane & 15;
  const int hiq  = lane >> 4;
  const int q0w  = blk * 64 + wave * 16;   // this wave's first query row

  __shared__ __align__(16) _Float16 kbuf[32][72];      // K block rows [key][d]
  __shared__ __align__(16) _Float16 vtbuf[64][40];     // V^T block [d][key]
  __shared__ __align__(16) _Float16 pbuf[4][16][40];   // per-wave P tile

  const size_t bhS = (size_t)bh * Ss;

  // Q fragments (16 rows x 64) resident in registers.
  const _Float16* qrow = qp + (bhS + q0w + lm) * DKk;
  const v16h qa0 = mk16(*(const v8h*)(qrow + hiq * 8),
                        *(const v8h*)(qrow + 16 + hiq * 8));
  const v16h qa1 = mk16(*(const v8h*)(qrow + 32 + hiq * 8),
                        *(const v8h*)(qrow + 48 + hiq * 8));

  v8f acc[4] = {};
  float m_i[8], l_i[8];
#pragma unroll
  for (int j = 0; j < 8; ++j) { m_i[j] = -1e30f; l_i[j] = 0.f; }

  const int kk   = t >> 2;            // staged key row (0..31)
  const int doff = (t & 3) * 16;      // 16-element d chunk
  const int jmax = blk * 64 + 64;     // causal bound for this query tile

  for (int j0 = 0; j0 < jmax; j0 += 32) {
    __syncthreads();
    {
      const _Float16* kg = kp + (bhS + j0 + kk) * DKk + doff;
      *(v8h*)&kbuf[kk][doff]     = *(const v8h*)kg;
      *(v8h*)&kbuf[kk][doff + 8] = *(const v8h*)(kg + 8);

      const _Float16* vg = vp + (bhS + j0 + kk) * DKk + doff;
      v8h v0 = *(const v8h*)vg, v1 = *(const v8h*)(vg + 8);
#pragma unroll
      for (int i = 0; i < 8; ++i) {            // transpose V into LDS
        vtbuf[doff + i][kk]     = v0[i];
        vtbuf[doff + 8 + i][kk] = v1[i];
      }
    }
    __syncthreads();

    // scores S = (Q K^T) / sqrt(64), two 16-wide key subtiles
    float pvals[2][8];
#pragma unroll
    for (int nt = 0; nt < 2; ++nt) {
      v16h bk0 = mk16(*(const v8h*)&kbuf[nt * 16 + lm][hiq * 8],
                      *(const v8h*)&kbuf[nt * 16 + lm][16 + hiq * 8]);
      v16h bk1 = mk16(*(const v8h*)&kbuf[nt * 16 + lm][32 + hiq * 8],
                      *(const v8h*)&kbuf[nt * 16 + lm][48 + hiq * 8]);
      v8f s = {};
      s = __builtin_amdgcn_wmma_f32_16x16x32_f16(false, qa0, false, bk0,
                                                 (short)0, s, false, false);
      s = __builtin_amdgcn_wmma_f32_16x16x32_f16(false, qa1, false, bk1,
                                                 (short)0, s, false, false);
      const int kcol = j0 + nt * 16 + lm;
#pragma unroll
      for (int j = 0; j < 8; ++j) {
        const int qr = q0w + j + 8 * hiq;
        pvals[nt][j] = (kcol <= qr) ? s[j] * 0.125f : -1e9f;   // causal
      }
    }

    // Online softmax (row reductions across the 16 lanes of each half-wave).
#pragma unroll
    for (int j = 0; j < 8; ++j) {
      float mrow = fmaxf(pvals[0][j], pvals[1][j]);
#pragma unroll
      for (int off = 1; off < 16; off <<= 1)
        mrow = fmaxf(mrow, __shfl_xor(mrow, off, 32));
      const float mnew  = fmaxf(m_i[j], mrow);
      const float alpha = __expf(m_i[j] - mnew);
      m_i[j] = mnew;
      const float p0 = __expf(pvals[0][j] - mnew);
      const float p1 = __expf(pvals[1][j] - mnew);
      float rs = p0 + p1;
#pragma unroll
      for (int off = 1; off < 16; off <<= 1)
        rs += __shfl_xor(rs, off, 32);
      l_i[j] = l_i[j] * alpha + rs;
#pragma unroll
      for (int tn = 0; tn < 4; ++tn) acc[tn][j] *= alpha;
      pbuf[wave][j + 8 * hiq][lm]      = (_Float16)p0;   // P -> A-frag layout
      pbuf[wave][j + 8 * hiq][16 + lm] = (_Float16)p1;
    }

    // O += P (16x32) x V (32x64)
    v16h pa = mk16(*(const v8h*)&pbuf[wave][lm][hiq * 8],
                   *(const v8h*)&pbuf[wave][lm][16 + hiq * 8]);
#pragma unroll
    for (int tn = 0; tn < 4; ++tn) {
      v16h bvf = mk16(*(const v8h*)&vtbuf[tn * 16 + lm][hiq * 8],
                      *(const v8h*)&vtbuf[tn * 16 + lm][16 + hiq * 8]);
      acc[tn] = __builtin_amdgcn_wmma_f32_16x16x32_f16(
          false, pa, false, bvf, (short)0, acc[tn], false, false);
    }
  }

  // Epilogue: normalize, write f16 context [B,S,H*DK] and attn_weights.
  float aw[8];
#pragma unroll
  for (int j = 0; j < 8; ++j) aw[j] = 0.f;
#pragma unroll
  for (int tn = 0; tn < 4; ++tn) {
#pragma unroll
    for (int j = 0; j < 8; ++j) {
      const float o = acc[tn][j] / l_i[j];
      aw[j] += o;
      const int qr = q0w + j + 8 * hiq;
      const int d  = tn * 16 + lm;
      ctx[((size_t)b * Ss + qr) * Dd + h * DKk + d] = (_Float16)o;
    }
  }
#pragma unroll
  for (int j = 0; j < 8; ++j) {
#pragma unroll
    for (int off = 1; off < 16; off <<= 1)
      aw[j] += __shfl_xor(aw[j], off, 32);
    if (lm == 0) {
      const int qr = q0w + j + 8 * hiq;
      attw[bhS + qr] = aw[j] * (1.0f / 64.0f);   // mean over d_k
    }
  }
}

// ---------------------------------------------------------------------------
extern "C" void kernel_launch(void* const* d_in, const int* in_sizes, int n_in,
                              void* d_out, int out_size, void* d_ws,
                              size_t ws_size, hipStream_t stream) {
  (void)in_sizes; (void)n_in; (void)out_size; (void)ws_size;
  const float* query = (const float*)d_in[0];
  const float* key_  = (const float*)d_in[1];
  const float* value = (const float*)d_in[2];
  // d_in[3] = attn_mask (causal tril) — reproduced structurally in-kernel.
  const float* Wq = (const float*)d_in[4];
  const float* bq = (const float*)d_in[5];
  const float* Wk = (const float*)d_in[6];
  const float* bk = (const float*)d_in[7];
  const float* Wv = (const float*)d_in[8];
  const float* bv = (const float*)d_in[9];
  const float* Wo = (const float*)d_in[10];
  const float* bo = (const float*)d_in[11];

  // Workspace layout (32 MB total).
  _Float16* qh  = (_Float16*)d_ws;                 // [B,H,S,DK]
  _Float16* kh  = qh + (size_t)Mrows * Dd;
  _Float16* vh  = kh + (size_t)Mrows * Dd;
  _Float16* ctx = vh + (size_t)Mrows * Dd;         // [B,S,D]

  float* out  = (float*)d_out;                     // [B,S,D]
  float* attw = out + (size_t)Mrows * Dd;          // [B,H,S]

  // 1) fused Q/K/V projections (z selects projection)
  gemm_qkvo<float, float, true>
      <<<dim3(Dd / 64, Mrows / 64, 3), dim3(128), 0, stream>>>(
          query, key_, value, Wq, Wk, Wv, bq, bk, bv, qh, kh, vh);

  // 2) causal flash attention
  flash_attn<<<dim3(Bb * Hh, Ss / 64), dim3(128), 0, stream>>>(qh, kh, vh, ctx,
                                                               attw);

  // 3) output projection (f16 activations, f32 weights)
  gemm_qkvo<_Float16, float, false>
      <<<dim3(Dd / 64, Mrows / 64, 1), dim3(128), 0, stream>>>(
          ctx, ctx, ctx, Wo, Wo, Wo, bo, bo, bo, out, out, out);
}
